// Net_996432413182
// MI455X (gfx1250) — matrix operand order repaired and checked by reading, not compiled, not run on hardware
//
#include <hip/hip_runtime.h>

#define N_NODES 50000
#define N_EDGES 1600000
#define HID 128
#define D_IN 15
#define C_DIM 8

typedef _Float16 half_t;
typedef __attribute__((ext_vector_type(16))) _Float16 v16h;
typedef __attribute__((ext_vector_type(8)))  _Float16 v8h;
typedef __attribute__((ext_vector_type(4)))  _Float16 v4h;
typedef __attribute__((ext_vector_type(8)))  float    v8f;

__device__ __forceinline__ float elu(float v) {
    return v > 0.0f ? v : (__expf(v) - 1.0f);
}

// ---------------- weight prep ----------------
// enc2 weight [K=128, N=128] f32 -> W^T [n][k] f16 (WMMA B-fragment friendly)
__global__ void prep_wt128(const float* __restrict__ w, half_t* __restrict__ wt) {
    int t = blockIdx.x * blockDim.x + threadIdx.x;
    if (t >= HID * HID) return;
    int n = t >> 7, k = t & 127;
    wt[n * HID + k] = (half_t)w[k * HID + n];
}

// conv weight [256,128]: emit (Wtop - Wbot)^T and Wbot^T in f16
__global__ void prep_conv_w(const float* __restrict__ w,
                            half_t* __restrict__ wt_amb,
                            half_t* __restrict__ wt_b) {
    int t = blockIdx.x * blockDim.x + threadIdx.x;
    if (t >= HID * HID) return;
    int n = t >> 7, k = t & 127;
    float top = w[k * HID + n];
    float bot = w[(k + HID) * HID + n];
    wt_amb[n * HID + k] = (half_t)(top - bot);
    wt_b[n * HID + k]   = (half_t)bot;
}

// fold eval-mode BatchNorm into scale/shift
__global__ void prep_bn(const float* __restrict__ g, const float* __restrict__ be,
                        const float* __restrict__ m, const float* __restrict__ v,
                        float* __restrict__ scale, float* __restrict__ shift) {
    int c = threadIdx.x;
    if (c < HID) {
        float s = g[c] * rsqrtf(v[c] + 1e-5f);
        scale[c] = s;
        shift[c] = be[c] - m[c] * s;
    }
}

// ---------------- encoder stage 1 (K=15, scalar) ----------------
__global__ void enc1_kernel(const float* __restrict__ x, const float* __restrict__ w,
                            const float* __restrict__ b, half_t* __restrict__ h1h) {
    int t = blockIdx.x * blockDim.x + threadIdx.x;
    if (t >= N_NODES * HID) return;
    int n = t >> 7, c = t & 127;
    float acc = b[c];
    const float* xr = x + n * D_IN;
#pragma unroll
    for (int k = 0; k < D_IN; ++k) acc = fmaf(xr[k], w[k * HID + c], acc);
    h1h[t] = (half_t)elu(acc);
}

// ---------------- WMMA fragment loads (contiguous k order chosen here, same for A and B) ----------------
// A 16x32 f16: lane<16 -> M=lane, K = kbase+0..15 contiguous
//              lane>=16 -> M=lane-16, K = kbase+16..31
__device__ __forceinline__ v16h load_a_frag(const half_t* __restrict__ base, int row,
                                            int kbase, int lane) {
    int koff = (lane >> 4) << 4;  // 0 or 16
    const half_t* p = base + (size_t)row * HID + kbase + koff;
    v8h lo = *(const v8h*)(p);
    v8h hi = *(const v8h*)(p + 8);
    v16h f;
#pragma unroll
    for (int i = 0; i < 8; ++i) { f[i] = lo[i]; f[i + 8] = hi[i]; }
    return f;
}

// B 32x16 f16 (stored as W^T[n][k]): lane<16 -> N=lane, K=kbase+0..15 contiguous
//                                    lane>=16 -> N=lane-16, K=kbase+16..31
__device__ __forceinline__ v16h load_b_frag(const half_t* __restrict__ wt, int col,
                                            int kbase, int lane) {
    int koff = (lane >> 4) << 4;  // 0 or 16
    return *(const v16h*)(wt + (size_t)col * HID + kbase + koff);
}

// ---------------- node-level GEMM: C[M,128] = Af16[M,128] @ W(128,128), f32 acc ----------------
// Block = 256 threads = 8 waves; block covers 16 rows, each wave one 16-col tile.
// TWO: second weight/output (A and B precompute share the feats loads).
// FUSE: bias+ELU epilogue and f16 copy emit (encoder stage 2).
template <int TWO, int FUSE>
__global__ void __launch_bounds__(256) node_gemm(const half_t* __restrict__ Ain,
                                                 const half_t* __restrict__ WT0,
                                                 const half_t* __restrict__ WT1,
                                                 const float* __restrict__ bias,
                                                 float* __restrict__ C0,
                                                 float* __restrict__ C1,
                                                 half_t* __restrict__ C0h) {
    const int wave = threadIdx.x >> 5;
    const int lane = threadIdx.x & 31;
    const int rowBase = blockIdx.x << 4;
    const int colBase = wave << 4;
    v8f acc0 = {};
    v8f acc1 = {};
    const int aRow = rowBase + (lane & 15);
    const int bCol = colBase + (lane & 15);
#pragma unroll
    for (int ks = 0; ks < 4; ++ks) {
        const int kbase = ks << 5;
        v16h a  = load_a_frag(Ain, aRow, kbase, lane);
        v16h b0 = load_b_frag(WT0, bCol, kbase, lane);
        acc0 = __builtin_amdgcn_wmma_f32_16x16x32_f16(false, a, false, b0,
                                                      (short)0, acc0, false, false);
        if (TWO) {
            v16h b1 = load_b_frag(WT1, bCol, kbase, lane);
            acc1 = __builtin_amdgcn_wmma_f32_16x16x32_f16(false, a, false, b1,
                                                          (short)0, acc1, false, false);
        }
    }
    // C/D layout: VGPR r -> M = rowBase + 8*(lane>=16) + r, N = colBase + (lane&15)
    const int m0 = rowBase + ((lane >> 4) << 3);
    const int n  = colBase + (lane & 15);
    const float bn = bias[n];
#pragma unroll
    for (int r = 0; r < 8; ++r) {
        const size_t idx = (size_t)(m0 + r) * HID + n;
        float v = acc0[r] + bn;
        if (FUSE) {
            v = elu(v);
            C0h[idx] = (half_t)v;
        }
        C0[idx] = v;
        if (TWO) C1[idx] = acc1[r];
    }
}

// ---------------- degree (shared by all 3 layers) ----------------
__global__ void deg_kernel(const int* __restrict__ row, float* __restrict__ deg) {
    int e = blockIdx.x * blockDim.x + threadIdx.x;
    if (e < N_EDGES)
        __hip_atomic_fetch_add(&deg[row[e]], 1.0f, __ATOMIC_RELAXED,
                               __HIP_MEMORY_SCOPE_AGENT);
}

// ---------------- per-edge: h = BN(ELU(A[row] + B[col])); sum[row] += h ----------------
// 32 threads per edge, 4 channels each (float4). A/B/sum are L2-resident (25.6 MB each).
__global__ void __launch_bounds__(256) edge_kernel(const float* __restrict__ A,
                                                   const float* __restrict__ B,
                                                   const int* __restrict__ row,
                                                   const int* __restrict__ col,
                                                   const float* __restrict__ scale,
                                                   const float* __restrict__ shift,
                                                   float* __restrict__ sum) {
    const long t = (long)blockIdx.x * blockDim.x + threadIdx.x;
    const int e = (int)(t >> 5);
    if (e >= N_EDGES) return;
    const int c0 = ((int)t & 31) << 2;
    const int r = row[e];
    const int c = col[e];
    const float4 a  = *(const float4*)(A + (size_t)r * HID + c0);
    const float4 b  = *(const float4*)(B + (size_t)c * HID + c0);
    const float4 sc = *(const float4*)(scale + c0);
    const float4 sh = *(const float4*)(shift + c0);
    float4 h;
    h.x = fmaf(elu(a.x + b.x), sc.x, sh.x);
    h.y = fmaf(elu(a.y + b.y), sc.y, sh.y);
    h.z = fmaf(elu(a.z + b.z), sc.z, sh.z);
    h.w = fmaf(elu(a.w + b.w), sc.w, sh.w);
    float* dst = sum + (size_t)r * HID + c0;
    __hip_atomic_fetch_add(dst + 0, h.x, __ATOMIC_RELAXED, __HIP_MEMORY_SCOPE_AGENT);
    __hip_atomic_fetch_add(dst + 1, h.y, __ATOMIC_RELAXED, __HIP_MEMORY_SCOPE_AGENT);
    __hip_atomic_fetch_add(dst + 2, h.z, __ATOMIC_RELAXED, __HIP_MEMORY_SCOPE_AGENT);
    __hip_atomic_fetch_add(dst + 3, h.w, __ATOMIC_RELAXED, __HIP_MEMORY_SCOPE_AGENT);
}

// ---------------- feats = sum/max(deg,1) + feats ; emit f16 copy ----------------
__global__ void finalize_kernel(const float* __restrict__ sum, const float* __restrict__ deg,
                                float* __restrict__ feats, half_t* __restrict__ feats16) {
    const long t = (long)blockIdx.x * blockDim.x + threadIdx.x;
    const int n = (int)(t >> 5);
    if (n >= N_NODES) return;
    const int c0 = ((int)t & 31) << 2;
    const float d = deg[n];
    const float inv = 1.0f / (d > 1.0f ? d : 1.0f);
    const size_t idx = (size_t)n * HID + c0;
    const float4 s = *(const float4*)(sum + idx);
    const float4 f = *(const float4*)(feats + idx);
    float4 v;
    v.x = fmaf(s.x, inv, f.x);
    v.y = fmaf(s.y, inv, f.y);
    v.z = fmaf(s.z, inv, f.z);
    v.w = fmaf(s.w, inv, f.w);
    *(float4*)(feats + idx) = v;
    v4h hv = {(half_t)v.x, (half_t)v.y, (half_t)v.z, (half_t)v.w};
    *(v4h*)(feats16 + idx) = hv;
}

// ---------------- fused output head: 128->32->16->8 with ELU, weights in LDS ----------------
__global__ void __launch_bounds__(256) head_kernel(const float* __restrict__ feats,
                                                   const float* __restrict__ w1,
                                                   const float* __restrict__ b1,
                                                   const float* __restrict__ w2,
                                                   const float* __restrict__ b2,
                                                   const float* __restrict__ w3,
                                                   const float* __restrict__ b3,
                                                   float* __restrict__ out) {
    __shared__ float W1s[HID * 32];
    __shared__ float W2s[32 * 16];
    __shared__ float W3s[16 * C_DIM];
    __shared__ float B1s[32], B2s[16], B3s[C_DIM];
    for (int i = threadIdx.x; i < HID * 32; i += blockDim.x) W1s[i] = w1[i];
    for (int i = threadIdx.x; i < 32 * 16; i += blockDim.x) W2s[i] = w2[i];
    for (int i = threadIdx.x; i < 16 * C_DIM; i += blockDim.x) W3s[i] = w3[i];
    if (threadIdx.x < 32) B1s[threadIdx.x] = b1[threadIdx.x];
    if (threadIdx.x < 16) B2s[threadIdx.x] = b2[threadIdx.x];
    if (threadIdx.x < C_DIM) B3s[threadIdx.x] = b3[threadIdx.x];
    __syncthreads();
    const int nIdx = blockIdx.x * blockDim.x + threadIdx.x;
    if (nIdx >= N_NODES) return;
    const float* fr = feats + (size_t)nIdx * HID;
    float o1[32];
#pragma unroll
    for (int o = 0; o < 32; ++o) o1[o] = B1s[o];
    for (int k = 0; k < HID; ++k) {
        const float f = fr[k];
#pragma unroll
        for (int o = 0; o < 32; ++o) o1[o] = fmaf(f, W1s[k * 32 + o], o1[o]);
    }
#pragma unroll
    for (int o = 0; o < 32; ++o) o1[o] = elu(o1[o]);
    float o2[16];
#pragma unroll
    for (int o = 0; o < 16; ++o) o2[o] = B2s[o];
#pragma unroll
    for (int k = 0; k < 32; ++k)
#pragma unroll
        for (int o = 0; o < 16; ++o) o2[o] = fmaf(o1[k], W2s[k * 16 + o], o2[o]);
#pragma unroll
    for (int o = 0; o < 16; ++o) o2[o] = elu(o2[o]);
    float o3[C_DIM];
#pragma unroll
    for (int o = 0; o < C_DIM; ++o) o3[o] = B3s[o];
#pragma unroll
    for (int k = 0; k < 16; ++k)
#pragma unroll
        for (int o = 0; o < C_DIM; ++o) o3[o] = fmaf(o2[k], W3s[k * C_DIM + o], o3[o]);
#pragma unroll
    for (int o = 0; o < C_DIM; ++o) out[(size_t)nIdx * C_DIM + o] = o3[o];
}

// ---------------- batch passthrough (second tuple output) ----------------
__global__ void batch_copy_f32(const int* __restrict__ batch, float* __restrict__ out) {
    int i = blockIdx.x * blockDim.x + threadIdx.x;
    if (i < N_NODES) out[i] = (float)batch[i];
}
__global__ void batch_copy_i64(const int* __restrict__ batch, int* __restrict__ out) {
    int i = blockIdx.x * blockDim.x + threadIdx.x;
    if (i < N_NODES) { out[2 * i] = batch[i]; out[2 * i + 1] = 0; }
}

extern "C" void kernel_launch(void* const* d_in, const int* in_sizes, int n_in,
                              void* d_out, int out_size, void* d_ws, size_t ws_size,
                              hipStream_t stream) {
    (void)in_sizes; (void)n_in; (void)ws_size;
    const float* x     = (const float*)d_in[0];
    const int*   ei    = (const int*)d_in[1];
    const int*   batch = (const int*)d_in[2];
    const float* e1w = (const float*)d_in[3];
    const float* e1b = (const float*)d_in[4];
    const float* e2w = (const float*)d_in[5];
    const float* e2b = (const float*)d_in[6];
    const float *cw[3], *cb[3], *cg[3], *cbe[3], *cm[3], *cv[3];
    for (int l = 0; l < 3; ++l) {
        cw[l]  = (const float*)d_in[7 + l * 6 + 0];
        cb[l]  = (const float*)d_in[7 + l * 6 + 1];
        cg[l]  = (const float*)d_in[7 + l * 6 + 2];
        cbe[l] = (const float*)d_in[7 + l * 6 + 3];
        cm[l]  = (const float*)d_in[7 + l * 6 + 4];
        cv[l]  = (const float*)d_in[7 + l * 6 + 5];
    }
    const float* o1w = (const float*)d_in[25];
    const float* o1b = (const float*)d_in[26];
    const float* o2w = (const float*)d_in[27];
    const float* o2b = (const float*)d_in[28];
    const float* o3w = (const float*)d_in[29];
    const float* o3b = (const float*)d_in[30];
    const int* rowp = ei;
    const int* colp = ei + N_EDGES;

    // workspace carve (all node buffers fit in the 192MB L2 together: ~115MB)
    char* ws = (char*)d_ws;
    size_t off = 0;
    auto carve = [&](size_t bytes) -> char* {
        char* p = ws + off;
        off += (bytes + 255) & ~(size_t)255;
        return p;
    };
    float*  feats   = (float*)carve((size_t)N_NODES * HID * 4);
    half_t* feats16 = (half_t*)carve((size_t)N_NODES * HID * 2);
    float*  Abuf    = (float*)carve((size_t)N_NODES * HID * 4);
    float*  Bbuf    = (float*)carve((size_t)N_NODES * HID * 4);
    char*   sumRaw  = carve((size_t)N_NODES * HID * 4);  // union: sum f32 / enc1 f16
    float*  sum     = (float*)sumRaw;
    half_t* h1h     = (half_t*)sumRaw;
    float*  deg     = (float*)carve((size_t)N_NODES * 4);
    half_t* wtE2    = (half_t*)carve(HID * HID * 2);
    half_t* wtAmb[3]; half_t* wtB[3]; float* bnScale[3]; float* bnShift[3];
    for (int l = 0; l < 3; ++l) {
        wtAmb[l]   = (half_t*)carve(HID * HID * 2);
        wtB[l]     = (half_t*)carve(HID * HID * 2);
        bnScale[l] = (float*)carve(HID * 4);
        bnShift[l] = (float*)carve(HID * 4);
    }

    // weight prep
    prep_wt128<<<(HID * HID + 255) / 256, 256, 0, stream>>>(e2w, wtE2);
    for (int l = 0; l < 3; ++l) {
        prep_conv_w<<<(HID * HID + 255) / 256, 256, 0, stream>>>(cw[l], wtAmb[l], wtB[l]);
        prep_bn<<<1, HID, 0, stream>>>(cg[l], cbe[l], cm[l], cv[l], bnScale[l], bnShift[l]);
    }

    // degree counts (identical across layers)
    hipMemsetAsync(deg, 0, N_NODES * 4, stream);
    deg_kernel<<<(N_EDGES + 255) / 256, 256, 0, stream>>>(rowp, deg);

    // encoder
    enc1_kernel<<<(N_NODES * HID + 255) / 256, 256, 0, stream>>>(x, e1w, e1b, h1h);
    node_gemm<0, 1><<<N_NODES / 16, 256, 0, stream>>>(h1h, wtE2, (half_t*)nullptr, e2b,
                                                      feats, (float*)nullptr, feats16);

    // 3 EdgeConv layers
    for (int l = 0; l < 3; ++l) {
        node_gemm<1, 0><<<N_NODES / 16, 256, 0, stream>>>(feats16, wtAmb[l], wtB[l], cb[l],
                                                          Abuf, Bbuf, (half_t*)nullptr);
        hipMemsetAsync(sum, 0, (size_t)N_NODES * HID * 4, stream);
        edge_kernel<<<(int)(((long)N_EDGES * 32 + 255) / 256), 256, 0, stream>>>(
            Abuf, Bbuf, rowp, colp, bnScale[l], bnShift[l], sum);
        finalize_kernel<<<(N_NODES * 32 + 255) / 256, 256, 0, stream>>>(sum, deg, feats,
                                                                        feats16);
    }

    // output head
    head_kernel<<<(N_NODES + 255) / 256, 256, 0, stream>>>(feats, o1w, o1b, o2w, o2b,
                                                           o3w, o3b, (float*)d_out);

    // batch passthrough (tuple output #2), handle either packing the harness uses
    if (out_size >= N_NODES * C_DIM + 2 * N_NODES) {
        batch_copy_i64<<<(N_NODES + 255) / 256, 256, 0, stream>>>(
            batch, (int*)((float*)d_out + (size_t)N_NODES * C_DIM));
    } else if (out_size >= N_NODES * C_DIM + N_NODES) {
        batch_copy_f32<<<(N_NODES + 255) / 256, 256, 0, stream>>>(
            batch, (float*)d_out + (size_t)N_NODES * C_DIM);
    }
}
